// InfillDiscriminatorPatchGAN_52072183496998
// MI455X (gfx1250) — matrix operand order repaired and checked by reading, not compiled, not run on hardware
//
#include <hip/hip_runtime.h>
#include <hip/hip_bf16.h>
#include <stdint.h>

typedef __bf16 bf16_t;
typedef __attribute__((ext_vector_type(16))) __bf16 v16bf;
typedef __attribute__((ext_vector_type(8)))  float  v8f;
typedef __attribute__((ext_vector_type(4)))  int    v4i;

#define IN_EPS 1e-5f

// ---- gfx1250 async global->LDS path (guarded; sync copy fallback) ----
#if defined(__HIP_DEVICE_COMPILE__) && defined(__gfx1250__) && \
    __has_builtin(__builtin_amdgcn_global_load_async_to_lds_b128)
#define HAVE_ASYNC_LDS 1
#else
#define HAVE_ASYNC_LDS 0
#endif

#if HAVE_ASYNC_LDS
// pointee typedefs carrying both the vector type and the address space;
// C-style cast is the only form clang accepts for cross-AS pointer casts.
typedef __attribute__((address_space(1))) v4i as1_v4i;   // global
typedef __attribute__((address_space(3))) v4i as3_v4i;   // LDS
__device__ __forceinline__ void async_cp16(const bf16_t* g, bf16_t* l) {
    __builtin_amdgcn_global_load_async_to_lds_b128(
        (as1_v4i*)g, (as3_v4i*)l, 0, 0);
}
#endif

__device__ __forceinline__ void wait_async_then_barrier() {
#if HAVE_ASYNC_LDS
#if __has_builtin(__builtin_amdgcn_s_wait_asynccnt)
    __builtin_amdgcn_s_wait_asynccnt(0);
#else
    asm volatile("s_wait_asynccnt 0" ::: "memory");
#endif
#endif
    __syncthreads();
}

__device__ __forceinline__ float lrelu(float v) { return v > 0.f ? v : 0.2f * v; }

// ------------------------------------------------------------------
// f32 -> bf16 repack (weights)
// ------------------------------------------------------------------
__global__ void cast_f32_bf16(const float* __restrict__ in, bf16_t* __restrict__ out, int n) {
    int i = blockIdx.x * blockDim.x + threadIdx.x;
    int st = gridDim.x * blockDim.x;
    for (; i < n; i += st) out[i] = (bf16_t)in[i];
}

// ------------------------------------------------------------------
// Layer 1: 1->64 channels, k=4, s=2, pad=1, + leaky ReLU, bf16 out.
// Memory-bound (C_in = 1): scalar FMAs, weights cached in LDS.
// ------------------------------------------------------------------
__global__ __launch_bounds__(256)
void conv1_lrelu(const float* __restrict__ x, const float* __restrict__ w1,
                 bf16_t* __restrict__ act1) {
    __shared__ float wsh[64 * 64];
    const int tid = threadIdx.x;
    for (int i = tid; i < 4096; i += 256) wsh[i] = w1[i];
    __syncthreads();

    unsigned o  = blockIdx.x * 256u + tid;     // < 2^25
    unsigned b  = o >> 24;
    unsigned co = (o >> 18) & 63u;
    unsigned s  = o & 262143u;
    int z = s >> 12, y = (s >> 6) & 63, xx = s & 63;
    const float* wc = &wsh[co * 64];

    float acc = 0.f;
#pragma unroll
    for (int kz = 0; kz < 4; ++kz) {
        int iz = 2 * z - 1 + kz;
        if ((unsigned)iz >= 128u) continue;
#pragma unroll
        for (int ky = 0; ky < 4; ++ky) {
            int iy = 2 * y - 1 + ky;
            if ((unsigned)iy >= 128u) continue;
            const float* row = x + (((size_t)b * 128 + iz) * 128 + iy) * 128;
#pragma unroll
            for (int kx = 0; kx < 4; ++kx) {
                int ix = 2 * xx - 1 + kx;
                if ((unsigned)ix >= 128u) continue;
                acc += row[ix] * wc[kz * 16 + ky * 4 + kx];
            }
        }
    }
    act1[o] = (bf16_t)lrelu(acc);
}

// ------------------------------------------------------------------
// Implicit-GEMM conv via WMMA bf16 -> f32, double-buffered LDS.
//   D[co][s] = sum_k W[co][k] * im2col(act)[k][s],  K = Cin*64.
// Block = 256 threads = 8 waves; block tile = 128 (Co) x 64 (spatial).
// Wave w owns rows co0+16w..+15 and 4 N-subtiles -> 4 accumulators,
// 4 v_wmma_f32_16x16x32_bf16 per K-chunk with the A fragment reused.
// Weights staged with global_load_async_to_lds_b128 (ping-pong buffers);
// im2col gather overlapped with the WMMA compute of the previous chunk.
// One s_wait_asynccnt + barrier per chunk.
// ------------------------------------------------------------------
__global__ __launch_bounds__(256)
void conv_wmma(const bf16_t* __restrict__ act, const bf16_t* __restrict__ wgt,
               float* __restrict__ out,
               int Cin, int inD, int outD, int S, int stride, int Co) {
    __shared__ __align__(16) bf16_t ldsA[2][128 * 32];   // 2 x 8 KB weights
    __shared__ __align__(16) bf16_t ldsB[2][64 * 32];    // 2 x 4 KB im2col
    __shared__ int ldsZ[64], ldsY[64], ldsX[64], ldsV[64];

    const int tid   = threadIdx.x;
    const int wave  = tid >> 5;
    const int lane  = tid & 31;
    const int b     = blockIdx.z;
    const int co0   = blockIdx.y * 128;
    const int tile0 = blockIdx.x * 64;
    const int K     = Cin * 64;
    const int chunks = Cin * 2;          // K/32
    const size_t inDD  = (size_t)inD * inD;
    const size_t inDDD = inDD * inD;

    // ---- one-time spatial decode for the 64 columns of this block
    if (tid < 64) {
        int s = tile0 + tid;
        int v = (s < S);
        int oDD = outD * outD;
        int z = s / oDD;
        int rem = s - z * oDD;
        int y = rem / outD;
        int x = rem - y * outD;
        ldsZ[tid] = z * stride - 1;      // pre-subtract pad, add kz later
        ldsY[tid] = y * stride - 1;
        ldsX[tid] = x * stride - 1;
        ldsV[tid] = v;
    }
    __syncthreads();

    // ---- per-thread gather geometry (fixed for whole kernel)
    const int gn   = tid >> 2;           // 0..63  column
    const int gkk0 = (tid & 3) * 8;      // 0,8,16,24 within chunk
    const int gz = ldsZ[gn], gy = ldsY[gn], gx = ldsX[gn], gv = ldsV[gn];

    // ---- fragment addressing (ISA 16-bit WMMA layouts)
    const int arow = (wave << 4) + (lane & 15);
    const int aklo = (lane & 16) ? 8 : 0;
    const int bcol = lane & 15;
    const int bks  = (lane & 16) ? 8 : 0;

    const int wrow  = tid >> 1;          // 0..127 weight row
    const int whalf = (tid & 1) << 4;    // 0 or 16

    const bf16_t* wsrc_base = wgt + (size_t)(co0 + wrow) * K + whalf;

    v8f acc[4] = {v8f{}, v8f{}, v8f{}, v8f{}};

    // ---- stage one K-chunk (32 wide) into LDS buffer `buf`
    auto stage = [&](int kc, int buf) {
        // weights: 128x32 bf16, 32 contiguous bytes per thread
        const bf16_t* src = wsrc_base + (kc << 5);
        bf16_t* dstA = &ldsA[buf][wrow * 32 + whalf];
#if HAVE_ASYNC_LDS
        async_cp16(src, dstA);
        async_cp16(src + 8, dstA + 8);
#else
        uint4 q0 = *(const uint4*)src;
        uint4 q1 = *(const uint4*)(src + 8);
        *(uint4*)dstA = q0;
        *(uint4*)(dstA + 8) = q1;
#endif
        // im2col: 64x32 bf16, 8 taps/thread, single ci per chunk
        const int ci    = kc >> 1;
        const int rbase = (kc & 1) << 5;
        const bf16_t* ap = act + ((size_t)(b * Cin + ci)) * inDDD;
        union { uint4 q; bf16_t h[8]; } pack;
#pragma unroll
        for (int u = 0; u < 8; ++u) {
            int r  = rbase + gkk0 + u;
            int kz = r >> 4, ky = (r >> 2) & 3, kx = r & 3;
            int iz = gz + kz, iy = gy + ky, ix = gx + kx;
            bf16_t val = (bf16_t)0.f;
            if (gv && (unsigned)iz < (unsigned)inD &&
                (unsigned)iy < (unsigned)inD && (unsigned)ix < (unsigned)inD) {
                val = ap[(size_t)iz * inDD + (size_t)iy * inD + ix];
            }
            pack.h[u] = val;
        }
        *(uint4*)&ldsB[buf][gn * 32 + gkk0] = pack.q;   // one ds_store_b128
    };

    stage(0, 0);
    wait_async_then_barrier();

    for (int kc = 0; kc < chunks; ++kc) {
        const int cur = kc & 1;
        if (kc + 1 < chunks) stage(kc + 1, cur ^ 1);   // overlap with WMMA below
        if (kc + 2 < chunks)                           // gfx1250 global_prefetch_b8
            __builtin_prefetch(wsrc_base + ((kc + 2) << 5), 0, 1);

        union { v16bf v; uint4 q[2]; } A;
        A.q[0] = *(const uint4*)&ldsA[cur][arow * 32 + aklo];
        A.q[1] = *(const uint4*)&ldsA[cur][arow * 32 + aklo + 16];
#pragma unroll
        for (int j = 0; j < 4; ++j) {
            union { v16bf v; uint4 q[2]; } B;
            const bf16_t* bp = &ldsB[cur][(j * 16 + bcol) * 32 + bks];
            B.q[0] = *(const uint4*)bp;
            B.q[1] = *(const uint4*)(bp + 16);
            acc[j] = __builtin_amdgcn_wmma_f32_16x16x32_bf16(
                false, A.v, false, B.v, (short)0, acc[j], false, false);
        }
        wait_async_then_barrier();
    }

    // ---- writeback (C layout: VGPR i -> M=i for lanes 0-15, M=i+8 for 16-31)
    const int mbase = (lane & 16) ? 8 : 0;
#pragma unroll
    for (int j = 0; j < 4; ++j) {
        int s = tile0 + j * 16 + (lane & 15);
        if (s < S) {
#pragma unroll
            for (int vi = 0; vi < 8; ++vi) {
                int co = co0 + (wave << 4) + mbase + vi;
                out[((size_t)b * Co + co) * S + s] = acc[j][vi];
            }
        }
    }
}

// ------------------------------------------------------------------
// InstanceNorm + leaky ReLU, f32 in -> bf16 out. One block per (b,c).
// ------------------------------------------------------------------
__global__ __launch_bounds__(256)
void inorm_lrelu(const float* __restrict__ pre, bf16_t* __restrict__ out, int S) {
    __shared__ float rs[256], rq[256];
    const int tid = threadIdx.x;
    const size_t base = (size_t)blockIdx.x * S;
    float s = 0.f, q = 0.f;
    for (int i = tid; i < S; i += 256) {
        float v = pre[base + i];
        s += v; q += v * v;
    }
    rs[tid] = s; rq[tid] = q;
    __syncthreads();
    for (int off = 128; off > 0; off >>= 1) {
        if (tid < off) { rs[tid] += rs[tid + off]; rq[tid] += rq[tid + off]; }
        __syncthreads();
    }
    float mean = rs[0] / S;
    float var  = rq[0] / S - mean * mean;
    float inv  = rsqrtf(var + IN_EPS);
    for (int i = tid; i < S; i += 256) {
        float v = (pre[base + i] - mean) * inv;
        out[base + i] = (bf16_t)lrelu(v);
    }
}

// ------------------------------------------------------------------
// Layer 4: 256->1, k=4, s=1, pad=1 (31^3 -> 30^3). Weights in LDS (32 KB).
// ------------------------------------------------------------------
__global__ __launch_bounds__(256)
void conv4(const bf16_t* __restrict__ act3, const bf16_t* __restrict__ w4,
           float* __restrict__ patch) {
    __shared__ bf16_t w4s[16384];
    const int tid = threadIdx.x;
    for (int i = tid; i < 16384; i += 256) w4s[i] = w4[i];
    __syncthreads();

    int o = blockIdx.x * 256 + tid;
    if (o >= 2 * 27000) return;
    int b = o / 27000;
    int s = o - b * 27000;
    int z = s / 900;
    int rem = s - z * 900;
    int y = rem / 30;
    int x = rem - y * 30;

    float acc = 0.f;
    for (int ci = 0; ci < 256; ++ci) {
        const bf16_t* ap = act3 + (size_t)(b * 256 + ci) * 29791;
        const bf16_t* wp = &w4s[ci * 64];
#pragma unroll
        for (int kz = 0; kz < 4; ++kz) {
            int iz = z - 1 + kz;
            if ((unsigned)iz >= 31u) continue;
#pragma unroll
            for (int ky = 0; ky < 4; ++ky) {
                int iy = y - 1 + ky;
                if ((unsigned)iy >= 31u) continue;
#pragma unroll
                for (int kx = 0; kx < 4; ++kx) {
                    int ix = x - 1 + kx;
                    if ((unsigned)ix >= 31u) continue;
                    acc += (float)ap[(iz * 31 + iy) * 31 + ix] *
                           (float)wp[kz * 16 + ky * 4 + kx];
                }
            }
        }
    }
    patch[o] = acc;
}

// ------------------------------------------------------------------
// Per-batch mean of 30^3 patch predictions -> out[b].
// ------------------------------------------------------------------
__global__ __launch_bounds__(256)
void batch_mean(const float* __restrict__ patch, float* __restrict__ out) {
    __shared__ float rs[256];
    const int tid = threadIdx.x;
    const int b = blockIdx.x;
    float s = 0.f;
    for (int i = tid; i < 27000; i += 256) s += patch[(size_t)b * 27000 + i];
    rs[tid] = s;
    __syncthreads();
    for (int off = 128; off > 0; off >>= 1) {
        if (tid < off) rs[tid] += rs[tid + off];
        __syncthreads();
    }
    if (tid == 0) out[b] = rs[0] / 27000.f;
}

// ------------------------------------------------------------------
extern "C" void kernel_launch(void* const* d_in, const int* in_sizes, int n_in,
                              void* d_out, int out_size, void* d_ws, size_t ws_size,
                              hipStream_t stream) {
    const float* x  = (const float*)d_in[0];
    const float* w1 = (const float*)d_in[1];
    const float* w2 = (const float*)d_in[2];
    const float* w3 = (const float*)d_in[3];
    const float* w4 = (const float*)d_in[4];
    float* out = (float*)d_out;

    char* p = (char*)d_ws;
    auto carve = [&](size_t bytes) {
        char* r = p;
        p += (bytes + 255) & ~(size_t)255;
        return r;
    };
    bf16_t* w2b   = (bf16_t*)carve((size_t)524288 * 2);     // 128*64*64
    bf16_t* w3b   = (bf16_t*)carve((size_t)2097152 * 2);    // 256*128*64
    bf16_t* w4b   = (bf16_t*)carve((size_t)16384 * 2);      // 1*256*64
    bf16_t* act1  = (bf16_t*)carve((size_t)33554432 * 2);   // 2*64*64^3
    float*  pre2  = (float*) carve((size_t)8388608 * 4);    // 2*128*32^3
    bf16_t* act2  = (bf16_t*)carve((size_t)8388608 * 2);
    float*  pre3  = (float*) carve((size_t)15252992 * 4);   // 2*256*31^3
    bf16_t* act3  = (bf16_t*)carve((size_t)15252992 * 2);
    float*  patch = (float*) carve((size_t)54000 * 4);      // 2*30^3

    // weight repack
    cast_f32_bf16<<<512, 256, 0, stream>>>(w2, w2b, 524288);
    cast_f32_bf16<<<2048, 256, 0, stream>>>(w3, w3b, 2097152);
    cast_f32_bf16<<<64, 256, 0, stream>>>(w4, w4b, 16384);

    // layer 1
    conv1_lrelu<<<131072, 256, 0, stream>>>(x, w1, act1);

    // layer 2: Cin=64, 64^3 -> 32^3, s=2, Co=128; K=4096
    conv_wmma<<<dim3(32768 / 64, 1, 2), 256, 0, stream>>>(
        act1, w2b, pre2, 64, 64, 32, 32768, 2, 128);
    inorm_lrelu<<<256, 256, 0, stream>>>(pre2, act2, 32768);

    // layer 3: Cin=128, 32^3 -> 31^3, s=1, Co=256; K=8192
    conv_wmma<<<dim3((29791 + 63) / 64, 2, 2), 256, 0, stream>>>(
        act2, w3b, pre3, 128, 32, 31, 29791, 1, 256);
    inorm_lrelu<<<512, 256, 0, stream>>>(pre3, act3, 29791);

    // layer 4 + reduction
    conv4<<<(54000 + 255) / 256, 256, 0, stream>>>(act3, w4b, patch);
    batch_mean<<<2, 256, 0, stream>>>(patch, out);
}
